// Conv1dGRU_43301860278887
// MI455X (gfx1250) — matrix operand order repaired and checked by reading, not compiled, not run on hardware
//
#include <hip/hip_runtime.h>
#include <hip/hip_bf16.h>

// ---------------------------------------------------------------------------
// Conv1d-GRU, 2 layers, fused WMMA step kernel for gfx1250 (MI455X).
//   T=128, N=16, CIN=COUT=64, W=256, K=3  -> all convs are GEMMs with
//   M=192 (3 gates x 64ch), Kdim=192 (64ci x 3 taps), Ncols=256 (spatial).
// v_wmma_f32_16x16x32_bf16 with f32 accumulators. bf16 operands staged into
// LDS with GLOBAL_LOAD_ASYNC_TO_LDS_B32 (ASYNCcnt), then repacked into an
// im2col tile whose address is affine in kk so every B fragment is two
// ds_load_b128 at immediate offsets. Fast v_rcp sigmoid; NT hints on
// single-use streams.
// ---------------------------------------------------------------------------

typedef __attribute__((ext_vector_type(16))) __bf16 v16bf;
typedef __attribute__((ext_vector_type(8)))  float  v8f;

union BFrag {
    unsigned short u[16];
    uint4          q[2];
    v16bf          v;
};
static_assert(sizeof(BFrag) == 32, "frag must be 32 bytes");

__device__ __forceinline__ unsigned short f2bf(float f) {
    unsigned u = __float_as_uint(f);
    u += 0x7FFFu + ((u >> 16) & 1u);   // round-to-nearest-even
    return (unsigned short)(u >> 16);
}

// Gate sigmoid: v_exp_f32 + v_rcp_f32 (no precise-division chain).
__device__ __forceinline__ float sigmoidf(float x) {
    return __builtin_amdgcn_rcpf(1.0f + __expf(-x));
}

// Async DMA: global (SGPR base + 32-bit byte voffset) -> LDS (byte address).
// Tracked by ASYNCcnt; completes into LDS without touching VGPRs.
__device__ __forceinline__ void async_ld_b32(const void* saddr, unsigned voff_bytes,
                                             unsigned lds_byte_addr) {
    asm volatile("global_load_async_to_lds_b32 %0, %1, %2"
                 :
                 : "v"(lds_byte_addr), "v"(voff_bytes), "s"(saddr)
                 : "memory");
}
__device__ __forceinline__ void wait_async0() {
    asm volatile("s_wait_asynccnt 0x0" ::: "memory");
}

// ---------------------------------------------------------------------------
// Weight packing: (192, 64, 3) f32 -> WMMA 16x16x32 bf16 A-fragments.
// A layout (contiguous k order, 16-bit A 16x32): lane l<16 holds row M=l with
// K {0..15} in v0..v7; lanes 16..31 hold row
// l-16 with K {16..31}.
// Fragment index: conv*72 + mt*6 + kc ; storage [frag][lane][16 halves].
// ---------------------------------------------------------------------------
__global__ __launch_bounds__(32)
void pack_weights_kernel(const float* __restrict__ w0,
                         const float* __restrict__ w1,
                         const float* __restrict__ w2,
                         const float* __restrict__ w3,
                         unsigned short* __restrict__ out) {
    int bid  = blockIdx.x;              // 0..287
    int conv = bid / 72;
    int rem  = bid % 72;
    int mt   = rem / 6;                 // M tile 0..11
    int kc   = rem % 6;                 // K chunk 0..5
    const float* w = (conv == 0) ? w0 : (conv == 1) ? w1 : (conv == 2) ? w2 : w3;

    int lane = threadIdx.x;
    int m    = mt * 16 + (lane & 15);
    int kblk = (lane < 16) ? 0 : 16;

    unsigned short vals[16];
#pragma unroll
    for (int d = 0; d < 8; ++d) {
        int kbase = kc * 32 + kblk + 2 * d;
#pragma unroll
        for (int h = 0; h < 2; ++h) {
            int kk = kbase + h;               // kk = ci*3 + tap, row stride 192
            vals[2 * d + h] = f2bf(w[m * 192 + kk]);
        }
    }
    unsigned short* dst = out + ((size_t)bid * 32 + lane) * 16;
#pragma unroll
    for (int i = 0; i < 16; ++i) dst[i] = vals[i];
}

// Combined per-layer bias table (256 floats):
//   [0..63]    b_i[r]+b_h[r]       [64..127]  b_i[z]+b_h[z]
//   [128..191] b_i[n]              [192..255] b_h[n]
__global__ __launch_bounds__(256)
void pack_bias_kernel(const float* __restrict__ b_i,
                      const float* __restrict__ b_h,
                      float* __restrict__ out) {
    int i = threadIdx.x;                 // 0..255
    int co = i & 63;
    float v;
    if (i < 128)      v = b_i[i] + b_h[i];        // r, z fused
    else if (i < 192) v = b_i[128 + co];          // n, i2h part
    else              v = b_h[128 + co];          // n, h2h part
    out[i] = v;
}

// f32 -> bf16 bulk convert (initial hidden-state mirror).
__global__ __launch_bounds__(256)
void f32_to_bf16_kernel(const float* __restrict__ src,
                        unsigned short* __restrict__ dst, int n) {
    int i = blockIdx.x * 256 + threadIdx.x;
    if (i < n) dst[i] = f2bf(src[i]);
}

// ---------------------------------------------------------------------------
// One GRU timestep for one layer.
// grid = (16 w-tiles, 16 samples), block = 128 (4 waves).
// Wave j owns gate tiles {j (r), j+4 (z), j+8 (n)} at its 16-column w tile.
//
// Raw stage: stride 40 halves, slot(ch,wl) = ch*40 + 1 + wl so the 16
// interior columns (wl=1..16 <-> gw=w0..w0+15) are 4B-aligned async b32
// targets; halos wl=0/17 patched with scalar stores.
// Im2col stage: [col][kk], row stride 200 halves (400B, 16B-multiple), so a
// lane's B fragment (kk = kc*32+koff+d, d=0..15) is 32 contiguous bytes at a
// compile-time offset -> two ds_load_b128, no address math.
// ---------------------------------------------------------------------------
#define SLOT(ch, wl) ((ch) * 40 + 1 + (wl))
#define IMROW 200

__global__ __launch_bounds__(128)
void gru_step_kernel(const float* __restrict__ in_f32,           // layer-0 input (or null)
                     const unsigned short* __restrict__ in_bf16, // layer-1 input (or null)
                     const float* __restrict__ h_in,             // f32 state (precise)
                     const unsigned short* __restrict__ h16_in,  // bf16 state mirror
                     float* __restrict__ h_out,
                     unsigned short* __restrict__ h16_out,
                     float* __restrict__ y_f32,                  // layer-1 output (or null)
                     unsigned short* __restrict__ y_bf16,        // layer-0 output (or null)
                     const unsigned short* __restrict__ wfrag_i2h,
                     const unsigned short* __restrict__ wfrag_h2h,
                     const float* __restrict__ bc) {             // combined bias table
    __shared__ alignas(16) unsigned short sX[64 * 40 + 24];      // raw input slice
    __shared__ alignas(16) unsigned short sH[64 * 40 + 24];      // raw state slice
    __shared__ alignas(16) unsigned short sXim[16 * IMROW + 8];  // im2col input
    __shared__ alignas(16) unsigned short sHim[16 * IMROW + 8];  // im2col state

    const int n   = blockIdx.y;
    const int w0  = blockIdx.x * 16;
    const int tid = threadIdx.x;
    const size_t rowBase = (size_t)n * 64 * 256;   // elements

    // ---- H interior: async bf16 DMA into LDS (64ch x 8 dword-pairs) ----
    for (int e = tid; e < 64 * 8; e += 128) {
        int ch = e >> 3, p = e & 7;
        unsigned lds  = (unsigned)(unsigned long long)&sH[SLOT(ch, 1 + 2 * p)];
        unsigned goff = (unsigned)((rowBase + (size_t)ch * 256 + w0 + 2 * p) * 2);
        async_ld_b32(h16_in, goff, lds);
    }
    // ---- X staging ----
    if (in_bf16) {
        // layer 1: async bf16 DMA for interior
        for (int e = tid; e < 64 * 8; e += 128) {
            int ch = e >> 3, p = e & 7;
            unsigned lds  = (unsigned)(unsigned long long)&sX[SLOT(ch, 1 + 2 * p)];
            unsigned goff = (unsigned)((rowBase + (size_t)ch * 256 + w0 + 2 * p) * 2);
            async_ld_b32(in_bf16, goff, lds);
        }
        // halo cells for X (wl=0 and wl=17), bounds-checked
        for (int e = tid; e < 64 * 2; e += 128) {
            int ch = e >> 1;
            int wl = (e & 1) ? 17 : 0;
            int gw = w0 - 1 + wl;
            unsigned short v = 0;
            if (gw >= 0 && gw < 256) v = in_bf16[rowBase + (size_t)ch * 256 + gw];
            sX[SLOT(ch, wl)] = v;
        }
    } else {
        // layer 0: synchronous f32 -> bf16 staging; x is read exactly once
        // over the whole problem -> non-temporal loads keep it out of L2.
        for (int e = tid; e < 64 * 18; e += 128) {
            int ch = e / 18, wl = e % 18;
            int gw = w0 - 1 + wl;
            unsigned short v = 0;
            if (gw >= 0 && gw < 256)
                v = f2bf(__builtin_nontemporal_load(
                        &in_f32[rowBase + (size_t)ch * 256 + gw]));
            sX[SLOT(ch, wl)] = v;
        }
    }
    // halo cells for H
    for (int e = tid; e < 64 * 2; e += 128) {
        int ch = e >> 1;
        int wl = (e & 1) ? 17 : 0;
        int gw = w0 - 1 + wl;
        unsigned short v = 0;
        if (gw >= 0 && gw < 256) v = h16_in[rowBase + (size_t)ch * 256 + gw];
        sH[SLOT(ch, wl)] = v;
    }

    wait_async0();       // all this wave's DMAs landed in LDS
    __syncthreads();     // all waves' DMAs + scalar patches visible

    // ---- repack raw -> im2col: cell (col, kk=ci*3+kr) = raw(ci, col+kr) ----
    // e decodes as (ci = e/48, kr = (e%48)/16, col = e%16): 64*3*16 = 3072
    // cells per buffer, each written exactly once (24 iters/thread).
    for (int e = tid; e < 64 * 3 * 16; e += 128) {
        int ci  = e / 48;
        int rem = e - ci * 48;
        int kr  = rem >> 4;        // 0..2
        int col = rem & 15;
        int kk  = ci * 3 + kr;
        sXim[col * IMROW + kk] = sX[SLOT(ci, col + kr)];
        sHim[col * IMROW + kk] = sH[SLOT(ci, col + kr)];
    }
    __syncthreads();

    const int wave = tid >> 5;
    const int lane = tid & 31;
    const int col  = lane & 15;
    const int koff = (lane < 16) ? 0 : 16;  // B layout: lane-half K split

    // Per-lane fragment base: 16B-aligned (col*400B + koff*2B in {0,32}).
    const unsigned short* bxBase = &sXim[col * IMROW + koff];
    const unsigned short* bhBase = &sHim[col * IMROW + koff];

    v8f accR  = (v8f)0.0f;   // r gate: i2h + h2h fused
    v8f accZ  = (v8f)0.0f;   // z gate: i2h + h2h fused
    v8f accNi = (v8f)0.0f;   // n gate, i2h part
    v8f accNh = (v8f)0.0f;   // n gate, h2h part (scaled by r later)

    auto ldA = [&](const unsigned short* base, int tile, int kc) -> v16bf {
        const BFrag* p =
            (const BFrag*)(base + ((size_t)(tile * 6 + kc) * 32 + lane) * 16);
        return p->v;   // two global_load_b128 per lane, L2-resident
    };

#pragma unroll
    for (int kc = 0; kc < 6; ++kc) {
        // ---- B fragments: 32 contiguous bytes at immediate offset ----
        BFrag bx, bh;
        bx.q[0] = *(const uint4*)(bxBase + kc * 32);
        bx.q[1] = *(const uint4*)(bxBase + kc * 32 + 8);
        bh.q[0] = *(const uint4*)(bhBase + kc * 32);
        bh.q[1] = *(const uint4*)(bhBase + kc * 32 + 8);
        // ---- i2h: r, z, ni ----
        v16bf aR = ldA(wfrag_i2h, wave,     kc);
        v16bf aZ = ldA(wfrag_i2h, wave + 4, kc);
        v16bf aN = ldA(wfrag_i2h, wave + 8, kc);
        accR  = __builtin_amdgcn_wmma_f32_16x16x32_bf16(false, aR, false, bx.v,
                                                        (short)0, accR, false, false);
        accZ  = __builtin_amdgcn_wmma_f32_16x16x32_bf16(false, aZ, false, bx.v,
                                                        (short)0, accZ, false, false);
        accNi = __builtin_amdgcn_wmma_f32_16x16x32_bf16(false, aN, false, bx.v,
                                                        (short)0, accNi, false, false);
        // ---- h2h: r, z, nh ----
        v16bf hR = ldA(wfrag_h2h, wave,     kc);
        v16bf hZ = ldA(wfrag_h2h, wave + 4, kc);
        v16bf hN = ldA(wfrag_h2h, wave + 8, kc);
        accR  = __builtin_amdgcn_wmma_f32_16x16x32_bf16(false, hR, false, bh.v,
                                                        (short)0, accR, false, false);
        accZ  = __builtin_amdgcn_wmma_f32_16x16x32_bf16(false, hZ, false, bh.v,
                                                        (short)0, accZ, false, false);
        accNh = __builtin_amdgcn_wmma_f32_16x16x32_bf16(false, hN, false, bh.v,
                                                        (short)0, accNh, false, false);
    }

    // ---- gate math + state update (C layout: VGPR v -> row v or v+8) ----
    const int rowAdd = (lane < 16) ? 0 : 8;
    const int w      = w0 + col;
#pragma unroll
    for (int v = 0; v < 8; ++v) {
        int co = wave * 16 + v + rowAdd;     // output channel 0..63
        float r  = sigmoidf(accR[v] + bc[co]);
        float z  = sigmoidf(accZ[v] + bc[64 + co]);
        float nc = sigmoidf(accNi[v] + bc[128 + co] +
                            r * (accNh[v] + bc[192 + co]));
        size_t idx = rowBase + (size_t)co * 256 + w;
        float hp = h_in[idx];
        float hn = (1.0f - z) * nc + z * hp;
        h_out[idx]   = hn;
        h16_out[idx] = f2bf(hn);
        if (y_bf16) y_bf16[idx] = f2bf(hn);                      // reused by layer 1
        else        __builtin_nontemporal_store(hn, &y_f32[idx]); // final, never re-read
    }
}

// ---------------------------------------------------------------------------
// Host-side launch: pack weights + biases, then per layer: init state mirror
// + 128 stream-ordered step kernels (launch boundary = global sync).
// ---------------------------------------------------------------------------
extern "C" void kernel_launch(void* const* d_in, const int* in_sizes, int n_in,
                              void* d_out, int out_size, void* d_ws, size_t ws_size,
                              hipStream_t stream) {
    (void)in_sizes; (void)n_in; (void)out_size; (void)ws_size;

    const float* x   = (const float*)d_in[0];   // (128,16,64,256)
    const float* h0  = (const float*)d_in[1];   // (2,16,64,256)
    const float* wi0 = (const float*)d_in[2];
    const float* bi0 = (const float*)d_in[3];
    const float* wh0 = (const float*)d_in[4];
    const float* bh0 = (const float*)d_in[5];
    const float* wi1 = (const float*)d_in[6];
    const float* bi1 = (const float*)d_in[7];
    const float* wh1 = (const float*)d_in[8];
    const float* bh1 = (const float*)d_in[9];
    float* out = (float*)d_out;

    const size_t slice = (size_t)16 * 64 * 256;        // per-t activation elems
    const size_t USH_PER_CONV = (size_t)72 * 32 * 16;  // 36864 bf16 per conv

    char* ws = (char*)d_ws;
    size_t off = 0;
    unsigned short* packW = (unsigned short*)(ws + off); off += 4 * USH_PER_CONV * 2;
    unsigned short* y0    = (unsigned short*)(ws + off); off += 128 * slice * 2;
    float* hA             = (float*)(ws + off);          off += slice * 4;
    float* hB             = (float*)(ws + off);          off += slice * 4;
    unsigned short* h16A  = (unsigned short*)(ws + off); off += slice * 2;
    unsigned short* h16B  = (unsigned short*)(ws + off); off += slice * 2;
    float* bcomb          = (float*)(ws + off);          off += 2 * 256 * 4;

    pack_weights_kernel<<<288, 32, 0, stream>>>(wi0, wh0, wi1, wh1, packW);
    pack_bias_kernel<<<1, 256, 0, stream>>>(bi0, bh0, bcomb);
    pack_bias_kernel<<<1, 256, 0, stream>>>(bi1, bh1, bcomb + 256);

    dim3 grid(16, 16);   // (w tiles, samples)

    for (int layer = 0; layer < 2; ++layer) {
        const float* hinit = h0 + (size_t)layer * slice;
        // seed the bf16 state mirror
        f32_to_bf16_kernel<<<(int)((slice + 255) / 256), 256, 0, stream>>>(
            hinit, h16A, (int)slice);

        const unsigned short* wI = packW + (size_t)(2 * layer + 0) * USH_PER_CONV;
        const unsigned short* wH = packW + (size_t)(2 * layer + 1) * USH_PER_CONV;
        const float* bC = bcomb + 256 * layer;

        float* ha = hA; float* hb = hB;
        unsigned short* ha16 = h16A; unsigned short* hb16 = h16B;
        for (int t = 0; t < 128; ++t) {
            const float* hi = (t == 0) ? hinit : ha;
            if (layer == 0) {
                gru_step_kernel<<<grid, 128, 0, stream>>>(
                    x + (size_t)t * slice, nullptr,
                    hi, ha16, hb, hb16,
                    nullptr, y0 + (size_t)t * slice,
                    wI, wH, bC);
            } else {
                gru_step_kernel<<<grid, 128, 0, stream>>>(
                    nullptr, y0 + (size_t)t * slice,
                    hi, ha16, hb, hb16,
                    out + (size_t)t * slice, nullptr,
                    wI, wH, bC);
            }
            float* tf = ha; ha = hb; hb = tf;
            unsigned short* ts = ha16; ha16 = hb16; hb16 = ts;
        }
    }
}
